// BinaryLinear_45586782880351
// MI455X (gfx1250) — matrix operand order repaired and checked by reading, not compiled, not run on hardware
//
#include <hip/hip_runtime.h>

#define M_BATCH 8192
#define IN_F    3072
#define HID     1024
#define CLS     10

typedef __attribute__((ext_vector_type(16))) __bf16 v16bf;
typedef __attribute__((ext_vector_type(8)))  __bf16 v8bf;
typedef __attribute__((ext_vector_type(8)))  float  v8f;

typedef int v4i __attribute__((vector_size(16)));
typedef __attribute__((address_space(1))) v4i gv4i;
typedef __attribute__((address_space(3))) v4i lv4i;
typedef __attribute__((address_space(1))) const __bf16 gbf16c;
typedef __attribute__((address_space(3))) __bf16 lbf16;

#if __has_builtin(__builtin_amdgcn_global_load_async_to_lds_b128)
#define HAVE_ASYNC 1
__device__ __forceinline__ void async_cp16(gbf16c* g, lbf16* l) {
  __builtin_amdgcn_global_load_async_to_lds_b128((gv4i*)g, (lv4i*)l, 0, 0);
}
#if __has_builtin(__builtin_amdgcn_s_wait_asynccnt)
#define WAIT_ASYNC(n) __builtin_amdgcn_s_wait_asynccnt(n)
#else
#define WAIT_ASYNC(n) asm volatile("s_wait_asynccnt %0" ::"i"(n) : "memory")
#endif
#else
#define HAVE_ASYNC 0
#define WAIT_ASYNC(n)
#endif

#define SHUF16(lo, hi) __builtin_shufflevector(lo, hi, 0,1,2,3,4,5,6,7,8,9,10,11,12,13,14,15)

// ---------------------------------------------------------------------------
// Main tiled GEMM: H = relu(A[MxK] * Bw[NxK]^T), bf16 in, bf16 out (f32 acc),
// plus deterministic per-block column sum / sum-of-squares partials for BN.
// Block = 256 threads (8 waves). Tile 128x128, K-step 32, double-buffered LDS
// fed by global_load_async_to_lds_b128 (ASYNCcnt). K/32 is even for all
// layers (96 or 32), so the k-loop is unrolled x2 with compile-time buffers.
// ---------------------------------------------------------------------------
__global__ __launch_bounds__(256)
void gemm_bin_relu_stats(const __bf16* __restrict__ A,
                         const __bf16* __restrict__ Bw,
                         __bf16* __restrict__ H,
                         float* __restrict__ gpsum,
                         float* __restrict__ gpsq,
                         int M, int N, int K)
{
  constexpr int LDT = 40;                      // padded LDS row stride (elems)
  constexpr int TILE_E = 128 * LDT;            // elems per buffer
  __shared__ __bf16 sA[2][TILE_E];
  __shared__ __bf16 sB[2][TILE_E];
  __shared__ float  ssum2[4][128];
  __shared__ float  ssq2[4][128];

  const int tid  = threadIdx.x;
  const int lane = tid & 31;
  const int wv   = tid >> 5;                   // wave 0..7
  const int wm   = wv >> 2;                    // 0..1  (M slice of 64)
  const int wn   = wv & 3;                     // 0..3  (N slice of 32)
  const int r    = lane & 15;
  const int hi   = lane >> 4;
  const int bn0  = blockIdx.x * 128;
  const int bm0  = blockIdx.y * 128;

  // loader mapping: thread -> (row 0..127, 16-elem half of the 32-elem K slab)
  const int lrow = tid >> 1;
  const int lcol = (tid & 1) * 16;
  const int lOff = lrow * LDT + lcol;

#if HAVE_ASYNC
  // Hoisted addrspace conversions: one-time casts, all loop addressing stays
  // in AS1/AS3 so the async ops get base+immediate addressing (no per-call
  // null-check cndmask from generic->AS3 casts).
  gbf16c* gA = (gbf16c*)(A  + (size_t)(bm0 + lrow) * K + lcol);
  gbf16c* gB = (gbf16c*)(Bw + (size_t)(bn0 + lrow) * K + lcol);
  lbf16*  lsA = (lbf16*)&sA[0][0];
  lbf16*  lsB = (lbf16*)&sB[0][0];
#else
  const __bf16* gA = A  + (size_t)(bm0 + lrow) * K + lcol;
  const __bf16* gB = Bw + (size_t)(bn0 + lrow) * K + lcol;
#endif

  v8f acc[4][2] = {};

  auto load_tile = [&](int kk, int bufsel) {
#if HAVE_ASYNC
    gbf16c* ga = gA + (size_t)kk * 32;
    gbf16c* gb = gB + (size_t)kk * 32;
    lbf16* la = lsA + bufsel * TILE_E + lOff;
    lbf16* lb = lsB + bufsel * TILE_E + lOff;
    async_cp16(ga,     la);
    async_cp16(ga + 8, la + 8);
    async_cp16(gb,     lb);
    async_cp16(gb + 8, lb + 8);
#else
    const __bf16* ga = gA + (size_t)kk * 32;
    const __bf16* gb = gB + (size_t)kk * 32;
    __bf16* la = &sA[bufsel][lOff];
    __bf16* lb = &sB[bufsel][lOff];
    uint4 a0 = *(const uint4*)ga;
    uint4 a1 = *(const uint4*)(ga + 8);
    uint4 b0 = *(const uint4*)gb;
    uint4 b1 = *(const uint4*)(gb + 8);
    *(uint4*)la = a0; *(uint4*)(la + 8) = a1;
    *(uint4*)lb = b0; *(uint4*)(lb + 8) = b1;
#endif
  };

  auto compute = [&](const __bf16* sa, const __bf16* sb) {
    // A frags: lane<16 holds K 0..7 / 16..23, lane>=16 holds K 8..15 / 24..31
    v16bf af[4];
#pragma unroll
    for (int i = 0; i < 4; ++i) {
      const int base = (wm * 64 + i * 16 + r) * LDT + hi * 8;
      v8bf lo = *(const v8bf*)(sa + base);
      v8bf h8 = *(const v8bf*)(sa + base + 16);
      af[i] = SHUF16(lo, h8);
    }
    // B frags: lane<16 = col n, K 0..7 / 16..23 ; lane>=16 = col n, K 8..15 / 24..31
    v16bf bv[2];
#pragma unroll
    for (int j = 0; j < 2; ++j) {
      const int base = (wn * 32 + j * 16 + r) * LDT + hi * 8;
      v8bf lo = *(const v8bf*)(sb + base);
      v8bf h8 = *(const v8bf*)(sb + base + 16);
      bv[j] = SHUF16(lo, h8);
    }
#pragma unroll
    for (int i = 0; i < 4; ++i)
#pragma unroll
      for (int j = 0; j < 2; ++j)
        acc[i][j] = __builtin_amdgcn_wmma_f32_16x16x32_bf16(
            false, af[i], false, bv[j], (short)0, acc[i][j], false, false);
  };

  const int nk = K >> 5;                       // even: 96 or 32
  load_tile(0, 0);
  for (int kk = 0; kk < nk; kk += 2) {
    load_tile(kk + 1, 1);                      // kk+1 < nk always (nk even)
    WAIT_ASYNC(4);                             // tile kk landed
    __syncthreads();
    compute(&sA[0][0], &sB[0][0]);
    __syncthreads();
    if (kk + 2 < nk) {
      load_tile(kk + 2, 0);
      WAIT_ASYNC(4);                           // tile kk+1 landed
    } else {
      WAIT_ASYNC(0);
    }
    __syncthreads();
    compute(&sA[1][0], &sB[1][0]);
    __syncthreads();
  }

  // Epilogue: relu, bf16 store, deterministic column stats.
  float sj[2] = {0.f, 0.f}, qj[2] = {0.f, 0.f};
#pragma unroll
  for (int i = 0; i < 4; ++i) {
#pragma unroll
    for (int j = 0; j < 2; ++j) {
      const size_t gcol = bn0 + wn * 32 + j * 16 + r;
#pragma unroll
      for (int e = 0; e < 8; ++e) {
        float v = acc[i][j][e];
        v = v > 0.f ? v : 0.f;
        sj[j] += v;
        qj[j] += v * v;
        const size_t grow = bm0 + wm * 64 + i * 16 + e + hi * 8;
        H[grow * (size_t)N + gcol] = (__bf16)v;
      }
    }
  }
  const int q4 = wm * 2 + hi;                  // unique contributor slot
#pragma unroll
  for (int j = 0; j < 2; ++j) {
    const int col = wn * 32 + j * 16 + r;
    ssum2[q4][col] = sj[j];
    ssq2[q4][col]  = qj[j];
  }
  __syncthreads();
  if (tid < 128) {
    const float s = ssum2[0][tid] + ssum2[1][tid] + ssum2[2][tid] + ssum2[3][tid];
    const float q = ssq2[0][tid]  + ssq2[1][tid]  + ssq2[2][tid]  + ssq2[3][tid];
    gpsum[(size_t)blockIdx.y * N + bn0 + tid] = s;
    gpsq [(size_t)blockIdx.y * N + bn0 + tid] = q;
  }
}

// ---------------------------------------------------------------------------
// Head layer: out[Mx10] = A[MxK] * Wb[16xK]^T (rows 10..15 of Wb are zero).
// One 16x16 WMMA tile per wave, A/B straight from global (Wb is L2-resident).
// ---------------------------------------------------------------------------
__global__ __launch_bounds__(256)
void gemm_head(const __bf16* __restrict__ A, const __bf16* __restrict__ Wb,
               float* __restrict__ out, int K)
{
  const int lane = threadIdx.x & 31;
  const int wv   = threadIdx.x >> 5;
  const int r    = lane & 15;
  const int hi   = lane >> 4;
  const int m0   = blockIdx.x * 128 + wv * 16;

  v8f acc = {};
  const __bf16* ga = A  + (size_t)(m0 + r) * K + hi * 8;
  const __bf16* gb = Wb + (size_t)r * K + hi * 8;
  for (int k = 0; k < K; k += 32) {
    v8bf a0 = *(const v8bf*)(ga + k);
    v8bf a1 = *(const v8bf*)(ga + k + 16);
    v8bf b0 = *(const v8bf*)(gb + k);
    v8bf b1 = *(const v8bf*)(gb + k + 16);
    v16bf av = SHUF16(a0, a1);
    v16bf bvv = SHUF16(b0, b1);
    acc = __builtin_amdgcn_wmma_f32_16x16x32_bf16(
        false, av, false, bvv, (short)0, acc, false, false);
  }
  if (r < CLS) {
#pragma unroll
    for (int e = 0; e < 8; ++e) {
      const int grow = m0 + e + hi * 8;
      out[(size_t)grow * CLS + r] = acc[e];
    }
  }
}

// ---------------------------------------------------------------------------
// Elementwise / prep kernels
// ---------------------------------------------------------------------------
__global__ void k_cvt_bf16(const float* __restrict__ x, __bf16* __restrict__ y, int n)
{
  const int i = (blockIdx.x * blockDim.x + threadIdx.x) * 4;
  if (i + 4 <= n) {
    float4 v = *(const float4*)(x + i);
    y[i + 0] = (__bf16)v.x; y[i + 1] = (__bf16)v.y;
    y[i + 2] = (__bf16)v.z; y[i + 3] = (__bf16)v.w;
  } else {
    for (int t = i; t < n; ++t) y[t] = (__bf16)x[t];
  }
}

__global__ void k_binarize(const float* __restrict__ w, __bf16* __restrict__ o,
                           int rows_valid, int cols, int total)
{
  const int i = blockIdx.x * blockDim.x + threadIdx.x;
  if (i < total) {
    const int row = i / cols;
    float v = 0.f;
    if (row < rows_valid) v = (w[i] >= 0.f) ? 1.f : -1.f;
    o[i] = (__bf16)v;
  }
}

__global__ void bn_finalize(const float* __restrict__ gpsum, const float* __restrict__ gpsq,
                            const float* __restrict__ gamma, const float* __restrict__ beta,
                            float* __restrict__ scale, float* __restrict__ shift,
                            int N, int nblk, float invM)
{
  const int i = blockIdx.x * blockDim.x + threadIdx.x;
  if (i < N) {
    float s = 0.f, q = 0.f;
    for (int b = 0; b < nblk; ++b) {
      s += gpsum[(size_t)b * N + i];
      q += gpsq [(size_t)b * N + i];
    }
    const float mean = s * invM;
    const float var  = q * invM - mean * mean;
    const float a    = gamma[i] * rsqrtf(var + 1e-5f);
    scale[i] = a;
    shift[i] = beta[i] - a * mean;
  }
}

__global__ void bn_apply(const __bf16* __restrict__ h,
                         const float* __restrict__ scale, const float* __restrict__ shift,
                         __bf16* __restrict__ y, int n)
{
  const int i = blockIdx.x * blockDim.x + threadIdx.x;
  if (i < n) {
    const int c = i & (HID - 1);
    y[i] = (__bf16)(scale[c] * (float)h[i] + shift[c]);
  }
}

// ---------------------------------------------------------------------------
extern "C" void kernel_launch(void* const* d_in, const int* in_sizes, int n_in,
                              void* d_out, int out_size, void* d_ws, size_t ws_size,
                              hipStream_t stream)
{
  (void)in_sizes; (void)n_in; (void)out_size; (void)ws_size;
  const float* x  = (const float*)d_in[0];
  const float* W1 = (const float*)d_in[1];
  const float* W2 = (const float*)d_in[2];
  const float* W3 = (const float*)d_in[3];
  const float* W4 = (const float*)d_in[4];
  const float* g1 = (const float*)d_in[5];
  const float* b1 = (const float*)d_in[6];
  const float* g2 = (const float*)d_in[7];
  const float* b2 = (const float*)d_in[8];
  const float* g3 = (const float*)d_in[9];
  const float* b3 = (const float*)d_in[10];
  float* out = (float*)d_out;

  char* ws = (char*)d_ws;
  size_t off = 0;
  auto wsalloc = [&](size_t bytes) -> void* {
    void* p = ws + off;
    off = (off + bytes + 255) & ~(size_t)255;
    return p;
  };
  const int nblk = M_BATCH / 128;
  __bf16* A0   = (__bf16*)wsalloc((size_t)M_BATCH * IN_F * 2);
  __bf16* AB   = (__bf16*)wsalloc((size_t)M_BATCH * HID * 2);
  __bf16* HB   = (__bf16*)wsalloc((size_t)M_BATCH * HID * 2);
  __bf16* Bw1  = (__bf16*)wsalloc((size_t)HID * IN_F * 2);
  __bf16* Bw2  = (__bf16*)wsalloc((size_t)HID * HID * 2);
  __bf16* Bw3  = (__bf16*)wsalloc((size_t)HID * HID * 2);
  __bf16* Wb4  = (__bf16*)wsalloc((size_t)16 * HID * 2);
  float* gpsum = (float*)wsalloc((size_t)nblk * HID * 4);
  float* gpsq  = (float*)wsalloc((size_t)nblk * HID * 4);
  float* scale = (float*)wsalloc(HID * 4);
  float* shift = (float*)wsalloc(HID * 4);

  // prep: bf16 activations + binarized bf16 weights (W layout == B^T layout)
  {
    const int n = M_BATCH * IN_F;
    k_cvt_bf16<<<(n / 4 + 255) / 256, 256, 0, stream>>>(x, A0, n);
  }
  k_binarize<<<(HID * IN_F + 255) / 256, 256, 0, stream>>>(W1, Bw1, HID, IN_F, HID * IN_F);
  k_binarize<<<(HID * HID + 255) / 256, 256, 0, stream>>>(W2, Bw2, HID, HID, HID * HID);
  k_binarize<<<(HID * HID + 255) / 256, 256, 0, stream>>>(W3, Bw3, HID, HID, HID * HID);
  k_binarize<<<(16 * HID + 255) / 256, 256, 0, stream>>>(W4, Wb4, CLS, HID, 16 * HID);

  const dim3 gg(HID / 128, M_BATCH / 128);
  const float invM = 1.0f / (float)M_BATCH;
  const int nh = M_BATCH * HID;

  auto layer = [&](const __bf16* Ain, int K, const __bf16* Bw,
                   const float* g, const float* b) {
    gemm_bin_relu_stats<<<gg, 256, 0, stream>>>(Ain, Bw, HB, gpsum, gpsq,
                                                M_BATCH, HID, K);
    bn_finalize<<<(HID + 255) / 256, 256, 0, stream>>>(gpsum, gpsq, g, b,
                                                       scale, shift, HID, nblk, invM);
    bn_apply<<<(nh + 255) / 256, 256, 0, stream>>>(HB, scale, shift, AB, nh);
  };

  layer(A0, IN_F, Bw1, g1, b1);
  layer(AB, HID,  Bw2, g2, b2);
  layer(AB, HID,  Bw3, g3, b3);

  gemm_head<<<M_BATCH / 128, 256, 0, stream>>>(AB, Wb4, out, HID);
}
